// TLSFunctionalGNN_82575041233135
// MI455X (gfx1250) — compile-verified, matrix operand order checked by reading, not run on hardware
//
#include <hip/hip_runtime.h>
#include <hip/hip_bf16.h>
#include <math.h>

// ---------------- problem constants (from reference) ----------------
#define BB    16
#define NPER  2048
#define NTOT  (BB * NPER)        // 32768
#define EE    524288
#define ESL   (EE + NTOT)        // edges + self loops = 557056
#define FIN   76
#define HID   128
#define HEADS 4
#define CH    32                 // HID/HEADS
#define K1    20
#define K2    5

// ---------------- WMMA types (CDNA5 gfx1250, wave32) ----------------
typedef __attribute__((ext_vector_type(16))) __bf16 v16bf;
typedef __attribute__((ext_vector_type(8)))  float  v8f;

union BF16Frag { v16bf v; __bf16 b[16]; };
union F32Acc   { v8f  v;  float e[8]; };

// ordered-uint mapping so unsigned atomicMax == float max
__device__ __forceinline__ unsigned f2u_ord(float f) {
    unsigned u = __float_as_uint(f);
    return (u & 0x80000000u) ? ~u : (u | 0x80000000u);
}
__device__ __forceinline__ float u2f_ord(unsigned u) {
    unsigned b = (u & 0x80000000u) ? (u & 0x7FFFFFFFu) : ~u;
    return __uint_as_float(b);
}
static inline unsigned host_f2u_ord(float f) {
    union { float f; unsigned u; } c; c.f = f;
    return (c.u & 0x80000000u) ? ~c.u : (c.u | 0x80000000u);
}

__device__ __forceinline__ float elu1(float x) { return x > 0.f ? x : (expf(x) - 1.f); }
__device__ __forceinline__ float lrelu(float x) { return x > 0.f ? x : 0.2f * x; }

// ---------------- utility fills ----------------
__global__ void fill_f32(float* p, long n, float v) {
    long i = (long)blockIdx.x * blockDim.x + threadIdx.x;
    if (i < n) p[i] = v;
}
__global__ void fill_u32(unsigned* p, long n, unsigned v) {
    long i = (long)blockIdx.x * blockDim.x + threadIdx.x;
    if (i < n) p[i] = v;
}

// ---------------- WMMA GEMM: C[M,N] = A[M,K] * B[K,N] (+bias, opt relu) -------
// one wave (32 threads) per 16x16 C tile; bf16 inputs, f32 accumulate.
// Out-of-range elements are handled branch-free: clamp index into range and
// multiply by a 0/1 mask (load is always valid -> no exec-mask branches).
__global__ void wmma_gemm_nn(const float* __restrict__ A, const float* __restrict__ Bm,
                             const float* __restrict__ bias, float* __restrict__ C,
                             int M, int N, int K, int act) {
    const int lane  = threadIdx.x & 31;
    const int col0  = blockIdx.x * 16;
    const int row0  = blockIdx.y * 16;
    const int half8 = (lane >> 4) * 8;
    const int rA    = row0 + (lane & 15);
    const int nB    = col0 + (lane & 15);

    F32Acc acc;
#pragma unroll
    for (int i = 0; i < 8; ++i) acc.e[i] = 0.f;

    const bool interior = (row0 + 16 <= M) && (col0 + 16 <= N);   // wave-uniform
    if (interior) {
        const float* Arow = A + (size_t)rA * K;
        const float* Bcol = Bm + nB;
        const int Kmain = K & ~31;
        for (int k0 = 0; k0 < Kmain; k0 += 32) {
            if (k0 + 32 < Kmain) {
                __builtin_prefetch(Arow + k0 + 32, 0, 1);
                __builtin_prefetch(Bcol + (size_t)(k0 + 32) * N, 0, 1);
            }
            BF16Frag a, b;
#pragma unroll
            for (int v = 0; v < 8; ++v) {
                const int kk = ((v & 4) << 2) + half8 + ((v & 3) << 1);
                const float2 pa = *(const float2*)(Arow + k0 + kk);   // contiguous pair
                a.b[2 * v]     = (__bf16)pa.x;
                a.b[2 * v + 1] = (__bf16)pa.y;
                b.b[2 * v]     = (__bf16)Bcol[(size_t)(k0 + kk) * N];
                b.b[2 * v + 1] = (__bf16)Bcol[(size_t)(k0 + kk + 1) * N];
            }
            acc.v = __builtin_amdgcn_wmma_f32_16x16x32_bf16(
                false, a.v, false, b.v, (short)0, acc.v, false, false);
        }
        if (Kmain < K) {                      // K tail (e.g. K=76), branch-free pad
            BF16Frag a, b;
#pragma unroll
            for (int v = 0; v < 8; ++v) {
                const int kk = ((v & 4) << 2) + half8 + ((v & 3) << 1);
#pragma unroll
                for (int bit = 0; bit < 2; ++bit) {
                    const int   kg = Kmain + kk + bit;
                    const int   kc = (kg < K) ? kg : (K - 1);
                    const float mk = (kg < K) ? 1.f : 0.f;
                    a.b[2 * v + bit] = (__bf16)(Arow[kc] * mk);
                    b.b[2 * v + bit] = (__bf16)(Bcol[(size_t)kc * N] * mk);
                }
            }
            acc.v = __builtin_amdgcn_wmma_f32_16x16x32_bf16(
                false, a.v, false, b.v, (short)0, acc.v, false, false);
        }
    } else {                                  // edge tiles: clamp + mask, no branches
        const int   rAc = (rA < M) ? rA : (M - 1);
        const int   nBc = (nB < N) ? nB : (N - 1);
        const float mA  = (rA < M) ? 1.f : 0.f;
        const float mB  = (nB < N) ? 1.f : 0.f;
        const float* Arow = A + (size_t)rAc * K;
        const float* Bcol = Bm + nBc;
        for (int k0 = 0; k0 < K; k0 += 32) {
            BF16Frag a, b;
#pragma unroll
            for (int v = 0; v < 8; ++v) {
                const int kk = ((v & 4) << 2) + half8 + ((v & 3) << 1);
#pragma unroll
                for (int bit = 0; bit < 2; ++bit) {
                    const int   kg = k0 + kk + bit;
                    const int   kc = (kg < K) ? kg : (K - 1);
                    const float mk = (kg < K) ? 1.f : 0.f;
                    a.b[2 * v + bit] = (__bf16)(Arow[kc] * (mA * mk));
                    b.b[2 * v + bit] = (__bf16)(Bcol[(size_t)kc * N] * (mB * mk));
                }
            }
            acc.v = __builtin_amdgcn_wmma_f32_16x16x32_bf16(
                false, a.v, false, b.v, (short)0, acc.v, false, false);
        }
    }

    if (nB < N) {
        float bi = bias ? bias[nB] : 0.f;
#pragma unroll
        for (int r = 0; r < 8; ++r) {
            int m = row0 + r + half8;
            if (m < M) {
                float o = acc.e[r] + bi;
                if (act == 1) o = fmaxf(o, 0.f);
                C[(size_t)m * N + nB] = o;
            }
        }
    }
}

// ---------------- batched WMMA GEMM: C = A^T * B ----------------
// A: (K x ldA) per batch (A^T[m,k] = A[k*ldA + m]); B: (K x N); C: (M x N)
// Loop-invariant lane guards turned into clamp + mask (branch-free inner loop).
__global__ void wmma_gemm_tn_b(const float* __restrict__ A, const float* __restrict__ Bm,
                               float* __restrict__ C, int M, int N, int K, int ldA,
                               long sA, long sB, long sC) {
    const int bt    = blockIdx.z;
    const int lane  = threadIdx.x & 31;
    const int col0  = blockIdx.x * 16;
    const int row0  = blockIdx.y * 16;
    const int half8 = (lane >> 4) * 8;
    const int rA    = row0 + (lane & 15);
    const int nB    = col0 + (lane & 15);
    const int   rAc = (rA < M) ? rA : (M - 1);
    const int   nBc = (nB < N) ? nB : (N - 1);
    const float mA  = (rA < M) ? 1.f : 0.f;
    const float mB  = (nB < N) ? 1.f : 0.f;
    const float* Acol = A + sA * bt + rAc;
    const float* Bcol = Bm + sB * bt + nBc;
    float* Cb         = C + sC * bt;

    F32Acc acc;
#pragma unroll
    for (int i = 0; i < 8; ++i) acc.e[i] = 0.f;

    for (int k0 = 0; k0 < K; k0 += 32) {     // K multiple of 32 here (2048)
        BF16Frag a, b;
#pragma unroll
        for (int v = 0; v < 8; ++v) {
            const int kk = ((v & 4) << 2) + half8 + ((v & 3) << 1);
#pragma unroll
            for (int bit = 0; bit < 2; ++bit) {
                const int kg = k0 + kk + bit;
                a.b[2 * v + bit] = (__bf16)(Acol[(size_t)kg * ldA] * mA);
                b.b[2 * v + bit] = (__bf16)(Bcol[(size_t)kg * N] * mB);
            }
        }
        acc.v = __builtin_amdgcn_wmma_f32_16x16x32_bf16(
            false, a.v, false, b.v, (short)0, acc.v, false, false);
    }
    if (nB < N) {
#pragma unroll
        for (int r = 0; r < 8; ++r) {
            int m = row0 + r + half8;
            if (m < M) Cb[(size_t)m * N + nB] = acc.e[r];
        }
    }
}

// ---------------- sparse GAT: per-node attention coefficients ----------------
__global__ void node_ad(const float* __restrict__ h, const float* __restrict__ a_src,
                        const float* __restrict__ a_dst, float* as_, float* ad_) {
    int t = blockIdx.x * blockDim.x + threadIdx.x;
    if (t >= NTOT * HEADS) return;
    int n = t >> 2, hh = t & 3;
    const float* hp = h + (size_t)n * HID + hh * CH;
    float s = 0.f, d = 0.f;
    for (int c = 0; c < CH; ++c) { float v = hp[c]; s += v * a_src[hh * CH + c]; d += v * a_dst[hh * CH + c]; }
    as_[t] = s; ad_[t] = d;
}

__global__ void edge_score_max(const int* __restrict__ src, const int* __restrict__ dst,
                               const float* __restrict__ as_, const float* __restrict__ ad_,
                               float* exbuf, unsigned* mu) {
    int t = blockIdx.x * blockDim.x + threadIdx.x;
    if (t >= ESL * HEADS) return;
    int e = t >> 2, h = t & 3;
    int s, d;
    if (e < EE) { s = src[e]; d = dst[e]; } else { s = d = e - EE; }
    float sc = lrelu(as_[s * HEADS + h] + ad_[d * HEADS + h]);
    exbuf[t] = sc;
    atomicMax(&mu[d * HEADS + h], f2u_ord(sc));
}

__global__ void edge_exp_den(const int* __restrict__ dst, float* exbuf,
                             const unsigned* __restrict__ mu, float* den) {
    int t = blockIdx.x * blockDim.x + threadIdx.x;
    if (t >= ESL * HEADS) return;
    int e = t >> 2, h = t & 3;
    int d = (e < EE) ? dst[e] : (e - EE);
    float ex = expf(exbuf[t] - u2f_ord(mu[d * HEADS + h]));
    exbuf[t] = ex;
    atomicAdd(&den[d * HEADS + h], ex);
}

__global__ void edge_scatter(const int* __restrict__ src, const int* __restrict__ dst,
                             const float* __restrict__ exbuf, const float* __restrict__ den,
                             const float* __restrict__ hlin, float* acc) {
    int t = blockIdx.x * blockDim.x + threadIdx.x;
    if (t >= ESL * HEADS) return;
    int e = t >> 2, h = t & 3;
    int s, d;
    if (e < EE) { s = src[e]; d = dst[e]; } else { s = d = e - EE; }
    float alpha = exbuf[t] / (den[d * HEADS + h] + 1e-16f);
    const float* hp = hlin + (size_t)s * HID + h * CH;
    float* op       = acc + (size_t)d * HID + h * CH;
    for (int c = 0; c < CH; ++c) atomicAdd(&op[c], alpha * hp[c]);
}

// bias + LayerNorm + ELU
__global__ void node_post(const float* __restrict__ acc, const float* __restrict__ bias,
                          const float* __restrict__ g, const float* __restrict__ be,
                          float* out) {
    int n = blockIdx.x * blockDim.x + threadIdx.x;
    if (n >= NTOT) return;
    const float* a = acc + (size_t)n * HID;
    float s = 0.f, s2 = 0.f;
    for (int c = 0; c < HID; ++c) { float t = a[c] + bias[c]; s += t; s2 += t * t; }
    float m   = s / (float)HID;
    float var = s2 / (float)HID - m * m;
    float inv = rsqrtf(var + 1e-5f);
    float* o  = out + (size_t)n * HID;
    for (int c = 0; c < HID; ++c) {
        float t  = a[c] + bias[c];
        float ln = (t - m) * inv * g[c] + be[c];
        o[c] = elu1(ln);
    }
}

// ---------------- diffpool pieces ----------------
// double softmax (PyG dense_diff_pool re-applies softmax) + entropy accumulation
__global__ void softmax2_ent(float* s, int rows, int kd, float* ent_acc) {
    int r = blockIdx.x * blockDim.x + threadIdx.x;
    if (r >= rows) return;
    float v[K1];
    float* p = s + (size_t)r * kd;
    float mx = -1e30f;
    for (int k = 0; k < kd; ++k) { v[k] = p[k]; mx = fmaxf(mx, v[k]); }
    float den = 0.f;
    for (int k = 0; k < kd; ++k) { v[k] = expf(v[k] - mx); den += v[k]; }
    for (int k = 0; k < kd; ++k) v[k] /= den;
    mx = -1e30f;
    for (int k = 0; k < kd; ++k) mx = fmaxf(mx, v[k]);
    den = 0.f;
    for (int k = 0; k < kd; ++k) { v[k] = expf(v[k] - mx); den += v[k]; }
    float ent = 0.f;
    for (int k = 0; k < kd; ++k) {
        float sv = v[k] / den;
        p[k] = sv;
        ent -= sv * logf(sv + 1e-15f);
    }
    atomicAdd(ent_acc, ent);
}

// ap[b,k,l] += s[src,k]*s[dst,l] over edges; also cross term of link loss
__global__ void ap_scatter(const float* __restrict__ s, const int* __restrict__ src,
                           const int* __restrict__ dst, float* ap, float* link_acc) {
    int e = blockIdx.x * blockDim.x + threadIdx.x;
    if (e >= EE) return;
    int si = src[e], di = dst[e];
    int b  = si >> 11;                       // NPER = 2048
    const float* ss = s + (size_t)si * K1;
    const float* sd = s + (size_t)di * K1;
    float a[K1], c[K1], cross = 0.f;
    for (int k = 0; k < K1; ++k) { a[k] = ss[k]; c[k] = sd[k]; cross += a[k] * c[k]; }
    atomicAdd(link_acc, -2.f * cross);
    float* apb = ap + (size_t)b * K1 * K1;
    for (int k = 0; k < K1; ++k) {
        float ak = a[k];
        for (int l = 0; l < K1; ++l) atomicAdd(&apb[k * K1 + l], ak * c[l]);
    }
}

// + sum_b ||S^T S||_F^2  (== ||S S^T||_F^2, avoids the N x N product)
__global__ void gram_link(const float* __restrict__ s, float* link_acc) {
    int b = blockIdx.x, t = threadIdx.x;
    if (t >= K1 * K1) return;
    int k = t / K1, l = t % K1;
    const float* sb = s + (size_t)b * NPER * K1;
    float g = 0.f;
    for (int n = 0; n < NPER; ++n) g += sb[n * K1 + k] * sb[n * K1 + l];
    atomicAdd(link_acc, g * g);
}

// ---------------- dense GAT (N=20, 4 heads), ELU fused ----------------
__global__ void dense_gat_attn(const float* __restrict__ hh, const float* __restrict__ asr,
                               const float* __restrict__ ads, const float* __restrict__ bias,
                               const float* __restrict__ adj, float* out) {
    int b = blockIdx.x, t = threadIdx.x;
    __shared__ float as_[K1 * HEADS], ad_[K1 * HEADS];
    const float* hb = hh + (size_t)b * K1 * HID;
    if (t < K1 * HEADS) {
        int j = t >> 2, h = t & 3;
        float sv = 0.f, dv = 0.f;
        for (int c = 0; c < CH; ++c) {
            float v = hb[j * HID + h * CH + c];
            sv += v * asr[h * CH + c]; dv += v * ads[h * CH + c];
        }
        as_[t] = sv; ad_[t] = dv;
    }
    __syncthreads();
    if (t < K1 * HEADS) {
        int i = t >> 2, h = t & 3;
        float e[K1], mx = -1e30f;
        for (int j = 0; j < K1; ++j) {
            float a  = (i == j) ? 1.f : adj[(size_t)b * K1 * K1 + i * K1 + j];
            float x  = ad_[i * HEADS + h] + as_[j * HEADS + h];
            float ev = (a == 0.f) ? -1e9f : lrelu(x);
            e[j] = ev; mx = fmaxf(mx, ev);
        }
        float den = 0.f;
        for (int j = 0; j < K1; ++j) { e[j] = expf(e[j] - mx); den += e[j]; }
        for (int c = 0; c < CH; ++c) {
            float o = 0.f;
            for (int j = 0; j < K1; ++j) o += e[j] * hb[j * HID + h * CH + c];
            float val = o / den + bias[h * CH + c];
            out[((size_t)b * K1 + i) * HID + h * CH + c] = elu1(val);
        }
    }
}

// ---------------- pool2 small dense pieces ----------------
__global__ void pool2_xp(const float* __restrict__ s2, const float* __restrict__ z2,
                         float* h3p) {
    int t = blockIdx.x * blockDim.x + threadIdx.x;
    if (t >= BB * K2 * HID) return;
    int b = t / (K2 * HID), rem = t % (K2 * HID);
    int k = rem / HID, f = rem % HID;
    float acc = 0.f;
    for (int n = 0; n < K1; ++n)
        acc += s2[((size_t)b * K1 + n) * K2 + k] * z2[((size_t)b * K1 + n) * HID + f];
    h3p[t] = acc;
}

__global__ void pool2_ap_link(const float* __restrict__ s2, const float* __restrict__ adj2,
                              float* adj3, float* link_acc) {
    int b = blockIdx.x, t = threadIdx.x;
    const float* sb = s2 + (size_t)b * K1 * K2;
    const float* ab = adj2 + (size_t)b * K1 * K1;
    if (t < K1 * K1) {                       // exact dense link loss at this level
        int n = t / K1, m = t % K1;
        float dt = 0.f;
        for (int k = 0; k < K2; ++k) dt += sb[n * K2 + k] * sb[m * K2 + k];
        float d = ab[t] - dt;
        atomicAdd(link_acc, d * d);
    }
    if (t < K2 * K2) {                       // ap3 = s^T adj2 s
        int k = t / K2, l = t % K2;
        float acc = 0.f;
        for (int n = 0; n < K1; ++n) {
            float snk = sb[n * K2 + k];
            float inner = 0.f;
            for (int m = 0; m < K1; ++m) inner += ab[n * K1 + m] * sb[m * K2 + l];
            acc += snk * inner;
        }
        adj3[(size_t)b * K2 * K2 + t] = acc;
    }
}

// ---------------- final dense GAT (1 head, ch=128, N=5), ELU fused ----------------
__global__ void gat3_attn(const float* __restrict__ hh, const float* __restrict__ asr,
                          const float* __restrict__ ads, const float* __restrict__ bias,
                          const float* __restrict__ adj, float* out) {
    int b = blockIdx.x, t = threadIdx.x;
    __shared__ float as_[K2], ad_[K2];
    const float* hb = hh + (size_t)b * K2 * HID;
    if (t < K2) {
        float s = 0.f, d = 0.f;
        for (int c = 0; c < HID; ++c) { float v = hb[t * HID + c]; s += v * asr[c]; d += v * ads[c]; }
        as_[t] = s; ad_[t] = d;
    }
    __syncthreads();
    if (t < HID) {
        for (int i = 0; i < K2; ++i) {
            float e[K2], mx = -1e30f;
            for (int j = 0; j < K2; ++j) {
                float a  = (i == j) ? 1.f : adj[(size_t)b * K2 * K2 + i * K2 + j];
                float x  = ad_[i] + as_[j];
                float ev = (a == 0.f) ? -1e9f : lrelu(x);
                e[j] = ev; mx = fmaxf(mx, ev);
            }
            float den = 0.f;
            for (int j = 0; j < K2; ++j) { e[j] = expf(e[j] - mx); den += e[j]; }
            float o = 0.f;
            for (int j = 0; j < K2; ++j) o += e[j] * hb[j * HID + t];
            out[(size_t)b * K2 * HID + i * HID + t] = elu1(o / den + bias[t]);
        }
    }
}

// ---------------- classifier head: mean-pool + MLP ----------------
__global__ void cls_head(const float* __restrict__ h3g, const float* __restrict__ W1,
                         const float* __restrict__ b1, const float* __restrict__ W2,
                         const float* __restrict__ b2, float* out) {
    int b = blockIdx.x, t = threadIdx.x;
    __shared__ float hg[HID], ls[64];
    if (t < HID) {
        float a = 0.f;
        for (int n = 0; n < K2; ++n) a += h3g[(size_t)b * K2 * HID + n * HID + t];
        hg[t] = a / (float)K2;
    }
    __syncthreads();
    if (t < 64) {
        float a = b1[t];
        for (int f = 0; f < HID; ++f) a += hg[f] * W1[f * 64 + t];
        ls[t] = fmaxf(a, 0.f);
    }
    __syncthreads();
    if (t < 2) {
        float a = b2[t];
        for (int u = 0; u < 64; ++u) a += ls[u] * W2[u * 2 + t];
        out[b * 2 + t] = a;
    }
}

__global__ void final_loss(const float* __restrict__ sc, float* out) {
    float l1 = sqrtf(fmaxf((float)EE + sc[0], 0.f)) /
               ((float)BB * (float)NPER * (float)NPER) + sc[1] / (float)NTOT;
    float l2 = sqrtf(fmaxf(sc[2], 0.f)) / (float)(BB * K1 * K1) + sc[3] / (float)(BB * K1);
    out[32] = l1 + l2;
}

// ================================================================
extern "C" void kernel_launch(void* const* d_in, const int* in_sizes, int n_in,
                              void* d_out, int out_size, void* d_ws, size_t ws_size,
                              hipStream_t stream) {
    (void)in_sizes; (void)n_in; (void)out_size; (void)ws_size;

    const float* x   = (const float*)d_in[0];
    const int*   ei  = (const int*)d_in[1];
    const int*   src = ei;
    const int*   dst = ei + EE;

    const float *g1a_W = (const float*)d_in[3],  *g1a_as = (const float*)d_in[4],
                *g1a_ad = (const float*)d_in[5], *g1a_b  = (const float*)d_in[6],
                *g1a_lg = (const float*)d_in[7], *g1a_lb = (const float*)d_in[8];
    const float *g1b_W = (const float*)d_in[9],  *g1b_as = (const float*)d_in[10],
                *g1b_ad = (const float*)d_in[11], *g1b_b = (const float*)d_in[12],
                *g1b_lg = (const float*)d_in[13], *g1b_lb = (const float*)d_in[14];
    const float *p1_aW1 = (const float*)d_in[15], *p1_ab1 = (const float*)d_in[16],
                *p1_aW2 = (const float*)d_in[17], *p1_ab2 = (const float*)d_in[18],
                *p1_eW  = (const float*)d_in[19], *p1_eb  = (const float*)d_in[20];
    const float *g2a_W = (const float*)d_in[21], *g2a_as = (const float*)d_in[22],
                *g2a_ad = (const float*)d_in[23], *g2a_b = (const float*)d_in[24];
    const float *g2b_W = (const float*)d_in[25], *g2b_as = (const float*)d_in[26],
                *g2b_ad = (const float*)d_in[27], *g2b_b = (const float*)d_in[28];
    const float *p2_aW1 = (const float*)d_in[29], *p2_ab1 = (const float*)d_in[30],
                *p2_aW2 = (const float*)d_in[31], *p2_ab2 = (const float*)d_in[32],
                *p2_eW  = (const float*)d_in[33], *p2_eb  = (const float*)d_in[34];
    const float *g3_W = (const float*)d_in[35], *g3_as = (const float*)d_in[36],
                *g3_ad = (const float*)d_in[37], *g3_b  = (const float*)d_in[38];
    const float *c_W1 = (const float*)d_in[39], *c_b1 = (const float*)d_in[40],
                *c_W2 = (const float*)d_in[41], *c_b2 = (const float*)d_in[42];

    // ---------------- workspace layout (floats) ----------------
    float* ws = (float*)d_ws;
    size_t off = 0;
    float* bufA = ws + off; off += (size_t)NTOT * HID;   // hlin / t1
    float* bufB = ws + off; off += (size_t)NTOT * HID;   // scatter accum / z
    float* hd   = ws + off; off += (size_t)NTOT * HID;   // layer outputs / hd
    float* as_  = ws + off; off += (size_t)NTOT * HEADS;
    float* ad_  = ws + off; off += (size_t)NTOT * HEADS;
    float* mbuf = ws + off; off += (size_t)NTOT * HEADS; // ordered-uint max
    float* den  = ws + off; off += (size_t)NTOT * HEADS;
    float* exb  = ws + off; off += (size_t)ESL * HEADS;
    float* sbuf = ws + off; off += (size_t)NTOT * K1;
    float* h2p  = ws + off; off += (size_t)BB * K1 * HID;
    float* hh   = ws + off; off += (size_t)BB * K1 * HID;
    float* h2p2 = ws + off; off += (size_t)BB * K1 * HID;
    float* adj2 = ws + off; off += (size_t)BB * K1 * K1;
    float* t2   = ws + off; off += (size_t)BB * K1 * HID;
    float* z2   = ws + off; off += (size_t)BB * K1 * HID;
    float* s2   = ws + off; off += (size_t)BB * K1 * K2;
    float* h3p  = ws + off; off += (size_t)BB * K2 * HID;
    float* adj3 = ws + off; off += (size_t)BB * K2 * K2;
    float* hh3  = ws + off; off += (size_t)BB * K2 * HID;
    float* h3g  = ws + off; off += (size_t)BB * K2 * HID;
    float* sc   = ws + off; off += 8;                    // [link1, ent1, link2, ent2]

    auto blks = [](long n, int b) { return (int)((n + b - 1) / b); };
    const unsigned NEGINF = host_f2u_ord(-1e30f);

    fill_f32<<<1, 32, 0, stream>>>(sc, 8, 0.f);

    // ---------------- sparse GAT layer helper ----------------
    auto sparse_gat = [&](const float* in, int fin, const float* W, const float* asr,
                          const float* ads, const float* bi, const float* lg,
                          const float* lb, float* outp) {
        dim3 g((HID + 15) / 16, (NTOT + 15) / 16);
        wmma_gemm_nn<<<g, 32, 0, stream>>>(in, W, nullptr, bufA, NTOT, HID, fin, 0);
        node_ad<<<blks((long)NTOT * HEADS, 256), 256, 0, stream>>>(bufA, asr, ads, as_, ad_);
        fill_u32<<<blks((long)NTOT * HEADS, 256), 256, 0, stream>>>((unsigned*)mbuf, (long)NTOT * HEADS, NEGINF);
        fill_f32<<<blks((long)NTOT * HEADS, 256), 256, 0, stream>>>(den, (long)NTOT * HEADS, 0.f);
        fill_f32<<<blks((long)NTOT * HID, 256), 256, 0, stream>>>(bufB, (long)NTOT * HID, 0.f);
        int ne = ESL * HEADS;
        edge_score_max<<<blks(ne, 256), 256, 0, stream>>>(src, dst, as_, ad_, exb, (unsigned*)mbuf);
        edge_exp_den<<<blks(ne, 256), 256, 0, stream>>>(dst, exb, (const unsigned*)mbuf, den);
        edge_scatter<<<blks(ne, 256), 256, 0, stream>>>(src, dst, exb, den, bufA, bufB);
        node_post<<<blks(NTOT, 256), 256, 0, stream>>>(bufB, bi, lg, lb, outp);
    };

    // GAT block 1a (FIN=76) and 1b (HID=128); hd = dense-batched node features
    sparse_gat(x,  FIN, g1a_W, g1a_as, g1a_ad, g1a_b, g1a_lg, g1a_lb, hd);
    sparse_gat(hd, HID, g1b_W, g1b_as, g1b_ad, g1b_b, g1b_lg, g1b_lb, hd);

    // ---------------- diffpool 1 (adj never materialized) ----------------
    {
        dim3 gh((HID + 15) / 16, (NTOT + 15) / 16);
        dim3 gs((K1 + 15) / 16, (NTOT + 15) / 16);
        wmma_gemm_nn<<<gh, 32, 0, stream>>>(hd, p1_aW1, p1_ab1, bufA, NTOT, HID, HID, 1);
        wmma_gemm_nn<<<gs, 32, 0, stream>>>(bufA, p1_aW2, p1_ab2, sbuf, NTOT, K1, HID, 0);
        wmma_gemm_nn<<<gh, 32, 0, stream>>>(hd, p1_eW, p1_eb, bufB, NTOT, HID, HID, 1);
        softmax2_ent<<<blks(NTOT, 256), 256, 0, stream>>>(sbuf, NTOT, K1, &sc[1]);
        fill_f32<<<blks(BB * K1 * K1, 256), 256, 0, stream>>>(adj2, BB * K1 * K1, 0.f);
        ap_scatter<<<blks(EE, 256), 256, 0, stream>>>(sbuf, src, dst, adj2, &sc[0]);
        gram_link<<<BB, 512, 0, stream>>>(sbuf, &sc[0]);
        // xp = s^T z per batch: (20 x 2048) * (2048 x 128)
        dim3 gt((HID + 15) / 16, (K1 + 15) / 16, BB);
        wmma_gemm_tn_b<<<gt, 32, 0, stream>>>(sbuf, bufB, h2p, K1, HID, NPER, K1,
                                              (long)NPER * K1, (long)NPER * HID, (long)K1 * HID);
    }

    // ---------------- dense GAT layers on pooled graph ----------------
    {
        dim3 g((HID + 15) / 16, (BB * K1 + 15) / 16);
        wmma_gemm_nn<<<g, 32, 0, stream>>>(h2p, g2a_W, nullptr, hh, BB * K1, HID, HID, 0);
        dense_gat_attn<<<BB, 128, 0, stream>>>(hh, g2a_as, g2a_ad, g2a_b, adj2, h2p2);
        wmma_gemm_nn<<<g, 32, 0, stream>>>(h2p2, g2b_W, nullptr, hh, BB * K1, HID, HID, 0);
        dense_gat_attn<<<BB, 128, 0, stream>>>(hh, g2b_as, g2b_ad, g2b_b, adj2, h2p);
    }

    // ---------------- diffpool 2 (tiny, exact) ----------------
    {
        dim3 gh((HID + 15) / 16, (BB * K1 + 15) / 16);
        dim3 gs((K2 + 15) / 16, (BB * K1 + 15) / 16);
        wmma_gemm_nn<<<gh, 32, 0, stream>>>(h2p, p2_aW1, p2_ab1, t2, BB * K1, HID, HID, 1);
        wmma_gemm_nn<<<gs, 32, 0, stream>>>(t2, p2_aW2, p2_ab2, s2, BB * K1, K2, HID, 0);
        wmma_gemm_nn<<<gh, 32, 0, stream>>>(h2p, p2_eW, p2_eb, z2, BB * K1, HID, HID, 1);
        softmax2_ent<<<blks(BB * K1, 256), 256, 0, stream>>>(s2, BB * K1, K2, &sc[3]);
        pool2_xp<<<blks(BB * K2 * HID, 256), 256, 0, stream>>>(s2, z2, h3p);
        pool2_ap_link<<<BB, 512, 0, stream>>>(s2, adj2, adj3, &sc[2]);
    }

    // ---------------- final GAT + classifier ----------------
    {
        dim3 g((HID + 15) / 16, (BB * K2 + 15) / 16);
        wmma_gemm_nn<<<g, 32, 0, stream>>>(h3p, g3_W, nullptr, hh3, BB * K2, HID, HID, 0);
        gat3_attn<<<BB, 128, 0, stream>>>(hh3, g3_as, g3_ad, g3_b, adj3, h3g);
        cls_head<<<BB, 128, 0, stream>>>(h3g, c_W1, c_b1, c_W2, c_b2, (float*)d_out);
        final_loss<<<1, 1, 0, stream>>>(sc, (float*)d_out);
    }
}
